// ComplexSelfAttention_3539053051944
// MI455X (gfx1250) — matrix operand order repaired and checked by reading, not compiled, not run on hardware
//
#include <hip/hip_runtime.h>
#include <hip/hip_bf16.h>
#include <stdint.h>

// ---------------------------------------------------------------------------
// Complex self-attention for MI455X (gfx1250).
//  - all matmuls via v_wmma_f32_16x16x32_bf16 (wave32)
//  - tile staging via GLOBAL_LOAD_ASYNC_TO_LDS_B128 (ASYNCcnt) with
//    double-buffered LDS in the GEMM so staging overlaps WMMA.
// Problem fits in 192MB L2 -> convert once to bf16 packed-complex form.
// ---------------------------------------------------------------------------

typedef __bf16 bf16;
typedef __attribute__((ext_vector_type(16))) __bf16 bf16x16;
typedef __attribute__((ext_vector_type(8)))  float  f32x8;

#define S_LEN   2048
#define F_DIM   1024
#define D_HEAD  64
#define MTOK    4096     // B * S
#define PK      2048     // packed complex feature dim (2*F head-dim concat)

union AF { bf16x16 v; unsigned u[8]; };
union U4 { uint4 q; bf16 h[8]; };

static __device__ __forceinline__ f32x8 wmma_bf16(const AF& a, const AF& b, f32x8 c) {
  // D = A(16x32 bf16) * B(32x16 bf16) + C(16x16 f32)
  return __builtin_amdgcn_wmma_f32_16x16x32_bf16(false, a.v, false, b.v,
                                                 (short)0, c, false, false);
}

// Async copy 16B from global directly into LDS (per-lane addresses).
// the LDS address comes from the vdst VGPR; low 32 bits of the flat shared pointer are the
// wave-relative LDS byte offset.  Tracked with ASYNCcnt.
static __device__ __forceinline__ void async_ld_b128(const void* gptr, void* lptr) {
  unsigned lds = (unsigned)(uintptr_t)lptr;
  asm volatile("global_load_async_to_lds_b128 %0, %1, off"
               :: "v"(lds), "v"(gptr) : "memory");
}

static __device__ __forceinline__ void wait_async0() {
#if __has_builtin(__builtin_amdgcn_s_wait_asynccnt)
  __builtin_amdgcn_s_wait_asynccnt(0);
#else
  asm volatile("s_wait_asynccnt 0" ::: "memory");
#endif
}

// ---------------------------------------------------------------------------
// Packing kernels (fp32 -> bf16, complex 2x2 block embedding)
// ---------------------------------------------------------------------------
__global__ __launch_bounds__(256)
void pack_x_kernel(const float* __restrict__ xr, const float* __restrict__ xi,
                   bf16* __restrict__ Xp) {
  size_t base = ((size_t)blockIdx.x * 256 + threadIdx.x) * 8;
  #pragma unroll
  for (int j = 0; j < 8; ++j) {
    size_t idx = base + j;
    size_t t = idx >> 11;              // token
    int    c = (int)(idx & (PK - 1));  // packed column
    float v = (c < F_DIM) ? xr[t * F_DIM + c] : xi[t * F_DIM + c - F_DIM];
    Xp[idx] = (bf16)v;
  }
}

// W~^T stored N-major: WT[n*2048 + k] = W~[k][n],
// W~ = [[Wr, Wi], [-Wi, Wr]]  (so X @ W~ = [yr | yi])
__global__ __launch_bounds__(256)
void pack_wT_kernel(const float* __restrict__ Wr, const float* __restrict__ Wi,
                    bf16* __restrict__ WT) {
  size_t base = ((size_t)blockIdx.x * 256 + threadIdx.x) * 8;
  #pragma unroll
  for (int j = 0; j < 8; ++j) {
    size_t idx = base + j;
    int n = (int)(idx >> 11);
    int k = (int)(idx & (PK - 1));
    int kr = k & (F_DIM - 1), nr = n & (F_DIM - 1);
    float v;
    if (k < F_DIM) v = (n < F_DIM) ?  Wr[(size_t)kr * F_DIM + nr] : Wi[(size_t)kr * F_DIM + nr];
    else           v = (n < F_DIM) ? -Wi[(size_t)kr * F_DIM + nr] : Wr[(size_t)kr * F_DIM + nr];
    WT[idx] = (bf16)v;
  }
}

// ---------------------------------------------------------------------------
// 64x64-tile WMMA GEMM:  C[M=4096][N=2048] = A[M][K=2048] @ B, with B given
// as BT[n][k] (N-major).  4 waves/block, each computes a 32x32 quadrant as
// 2x2 wmma tiles.  LDS double-buffered; staging uses async-to-LDS so tile
// k+1 streams in while tile k feeds the WMMAs.
// MODE 0: bf16 packed output (+bias).  MODE 1: fp32 split real/imag (+bias).
// ---------------------------------------------------------------------------
template<int MODE>
__global__ __launch_bounds__(128)
void gemm64x64(const bf16* __restrict__ A, const bf16* __restrict__ BT,
               const float* __restrict__ br, const float* __restrict__ bi,
               void* __restrict__ out) {
  const int K = PK, N = PK;
  __shared__ __align__(16) bf16 As[2][64 * 32];
  __shared__ __align__(16) bf16 Bs[2][64 * 32];
  const int tid = threadIdx.x;
  const int m0 = blockIdx.y * 64, n0 = blockIdx.x * 64;
  const int wid = tid >> 5, lane = tid & 31, l16 = lane & 15, hf = lane >> 4;
  const int wm = (wid >> 1) * 32, wn = (wid & 1) * 32;

  const bf16* gA = A  + (size_t)m0 * K;
  const bf16* gB = BT + (size_t)n0 * K;

  // staging geometry: 256 16B-chunks per 64x32 tile, 2 per thread
  const int c0  = tid * 2;
  const int r0  = c0 >> 2,       cl0 = (c0 & 3) * 8;
  const int r1  = (c0 + 1) >> 2, cl1 = ((c0 + 1) & 3) * 8;

  f32x8 acc[2][2] = {};

  // prologue: stage k0 = 0 into buffer 0
  async_ld_b128(&gA[(size_t)r0 * K + cl0], &As[0][r0 * 32 + cl0]);
  async_ld_b128(&gB[(size_t)r0 * K + cl0], &Bs[0][r0 * 32 + cl0]);
  async_ld_b128(&gA[(size_t)r1 * K + cl1], &As[0][r1 * 32 + cl1]);
  async_ld_b128(&gB[(size_t)r1 * K + cl1], &Bs[0][r1 * 32 + cl1]);
  wait_async0();
  __syncthreads();

  int buf = 0;
  for (int k0 = 0; k0 < K; k0 += 32) {
    if (k0 + 32 < K) {   // stage next tile into the other buffer (async)
      const bf16* nA = gA + k0 + 32;
      const bf16* nB = gB + k0 + 32;
      async_ld_b128(&nA[(size_t)r0 * K + cl0], &As[buf ^ 1][r0 * 32 + cl0]);
      async_ld_b128(&nB[(size_t)r0 * K + cl0], &Bs[buf ^ 1][r0 * 32 + cl0]);
      async_ld_b128(&nA[(size_t)r1 * K + cl1], &As[buf ^ 1][r1 * 32 + cl1]);
      async_ld_b128(&nB[(size_t)r1 * K + cl1], &Bs[buf ^ 1][r1 * 32 + cl1]);
    }

    const bf16* lA = As[buf];
    const bf16* lB = Bs[buf];
    AF a[2], b[2];
    #pragma unroll
    for (int mt = 0; mt < 2; ++mt) {
      int row = wm + mt * 16 + l16;
      #pragma unroll
      for (int v = 0; v < 8; ++v) {
        int kk = (v < 4) ? (hf * 8 + 2 * v) : (16 + hf * 8 + 2 * (v - 4));
        a[mt].u[v] = *(const unsigned*)&lA[row * 32 + kk];
      }
    }
    #pragma unroll
    for (int nt = 0; nt < 2; ++nt) {
      int row = wn + nt * 16 + l16;
      #pragma unroll
      for (int v = 0; v < 8; ++v) {
        int kk = (v < 4) ? (hf * 8 + 2 * v) : (16 + hf * 8 + 2 * (v - 4));
        b[nt].u[v] = *(const unsigned*)&lB[row * 32 + kk];
      }
    }
    #pragma unroll
    for (int mt = 0; mt < 2; ++mt)
      #pragma unroll
      for (int nt = 0; nt < 2; ++nt)
        acc[mt][nt] = wmma_bf16(a[mt], b[nt], acc[mt][nt]);

    wait_async0();      // next buffer fully in LDS
    __syncthreads();    // all waves done reading current buffer
    buf ^= 1;
  }

  #pragma unroll
  for (int mt = 0; mt < 2; ++mt)
    #pragma unroll
    for (int nt = 0; nt < 2; ++nt)
      #pragma unroll
      for (int r = 0; r < 8; ++r) {
        int m = m0 + wm + mt * 16 + r + 8 * hf;   // C-layout: M = r (+8 for hi half)
        int n = n0 + wn + nt * 16 + l16;          // N = lane%16
        float v = acc[mt][nt][r] + ((n < F_DIM) ? br[n] : bi[n - F_DIM]);
        if (MODE == 0) {
          ((bf16*)out)[(size_t)m * N + n] = (bf16)v;
        } else {
          int part = (n >= F_DIM) ? 1 : 0;
          int col  = n & (F_DIM - 1);
          ((float*)out)[(size_t)part * MTOK * F_DIM + (size_t)m * F_DIM + col] = v;
        }
      }
}

// ---------------------------------------------------------------------------
// Flash-style attention per (batch, head, 64-row q tile).  4 waves x 16 q-rows.
// scores = (Qr.Kr + Qi.Ki) * scale  -> one K=128 WMMA chain over [Qr|Qi].
// K-tile staged with async-to-LDS; V transposed manually (scatter).
// ---------------------------------------------------------------------------
__global__ __launch_bounds__(128)
void attn64(const bf16* __restrict__ Qp, const bf16* __restrict__ Kp,
            const bf16* __restrict__ Vp, const float* __restrict__ mask,
            bf16* __restrict__ Cp) {
  __shared__ __align__(16) bf16 Ks [64 * 128];   // [key][0..63 re | 64..127 im]
  __shared__ __align__(16) bf16 VsR[64 * 64];    // [dim][key]  (transposed)
  __shared__ __align__(16) bf16 VsI[64 * 64];
  __shared__ __align__(16) bf16 Ps [4][16 * 64]; // per-wave P tile

  const int tid = threadIdx.x, wid = tid >> 5, lane = tid & 31;
  const int l16 = lane & 15, hf = lane >> 4;
  const int bh = blockIdx.x, b = bh >> 4, h = bh & 15;
  const int q0 = blockIdx.y * 64;
  const int tb = b * S_LEN;
  const int cr = h * D_HEAD, ci = F_DIM + h * D_HEAD;
  const float scale = 0.125f;   // 1/sqrt(64)

  // Q A-fragments for this wave's 16 rows, K-depth 128 = 4 frags, kept in VGPRs
  AF qa[4];
  {
    const bf16* qrow = Qp + (size_t)(tb + q0 + wid * 16 + l16) * PK;
    #pragma unroll
    for (int f = 0; f < 4; ++f)
      #pragma unroll
      for (int v = 0; v < 8; ++v) {
        int kk   = (v < 4) ? (hf * 8 + 2 * v) : (16 + hf * 8 + 2 * (v - 4));
        int dcat = f * 32 + kk;                       // 0..127 over [re|im]
        int col  = (dcat < 64) ? (cr + dcat) : (ci + dcat - 64);
        qa[f].u[v] = *(const unsigned*)&qrow[col];
      }
  }

  f32x8 aR[4] = {}, aI[4] = {};
  float rm[8], rs[8];
  #pragma unroll
  for (int r = 0; r < 8; ++r) { rm[r] = -1e30f; rs[r] = 0.f; }

  for (int kb = 0; kb < S_LEN / 64; ++kb) {
    const int kbase = kb * 64;
    // stage K tile via async-to-LDS (natural layout == B-fragment n-major)
    #pragma unroll
    for (int i = 0; i < 8; ++i) {
      int c = tid + i * 128;                 // 1024 16B chunks
      int key = c >> 4, sub = c & 15;
      const bf16* krow = Kp + (size_t)(tb + kbase + key) * PK;
      int col = (sub < 8) ? (cr + sub * 8) : (ci + (sub - 8) * 8);
      int dst = key * 128 + ((sub < 8) ? sub * 8 : 64 + (sub - 8) * 8);
      async_ld_b128(&krow[col], &Ks[dst]);
    }
    // stage V tiles transposed ([dim][key]) for B-fragments of P@V
    #pragma unroll
    for (int i = 0; i < 4; ++i) {
      int c = tid + i * 128;                 // 512 chunks
      int key = c >> 3, sub = c & 7;
      const bf16* vrow = Vp + (size_t)(tb + kbase + key) * PK;
      U4 ur, ui;
      ur.q = *(const uint4*)&vrow[cr + sub * 8];
      ui.q = *(const uint4*)&vrow[ci + sub * 8];
      #pragma unroll
      for (int j = 0; j < 8; ++j) {
        VsR[(sub * 8 + j) * 64 + key] = ur.h[j];
        VsI[(sub * 8 + j) * 64 + key] = ui.h[j];
      }
    }
    wait_async0();
    __syncthreads();

    // ---- scores: 16 q-rows x 64 keys ----
    float Sv[4][8];
    #pragma unroll
    for (int kt = 0; kt < 4; ++kt) {
      f32x8 s = {};
      int keyrow = kt * 16 + l16;            // B-frag: n = key
      #pragma unroll
      for (int f = 0; f < 4; ++f) {
        AF bk;
        #pragma unroll
        for (int v = 0; v < 8; ++v)
          bk.u[v] = *(const unsigned*)&Ks[keyrow * 128 + f * 32 + ((v < 4) ? (hf * 8 + 2 * v) : (16 + hf * 8 + 2 * (v - 4)))];
        s = wmma_bf16(qa[f], bk, s);
      }
      float madd = (1.0f - mask[b * S_LEN + kbase + kt * 16 + l16]) * (-1e9f);
      #pragma unroll
      for (int r = 0; r < 8; ++r) Sv[kt][r] = s[r] * scale + madd;
    }

    // ---- online softmax (row stats split across the two 16-lane halves) ----
    #pragma unroll
    for (int r = 0; r < 8; ++r) {
      float mx = Sv[0][r];
      #pragma unroll
      for (int kt = 1; kt < 4; ++kt) mx = fmaxf(mx, Sv[kt][r]);
      #pragma unroll
      for (int off = 8; off >= 1; off >>= 1) mx = fmaxf(mx, __shfl_xor(mx, off, 32));
      float nm = fmaxf(rm[r], mx);
      float alpha = __expf(rm[r] - nm);
      rm[r] = nm;
      float ls = 0.f;
      #pragma unroll
      for (int kt = 0; kt < 4; ++kt) {
        float p = __expf(Sv[kt][r] - nm);
        Sv[kt][r] = p;
        ls += p;
      }
      #pragma unroll
      for (int off = 8; off >= 1; off >>= 1) ls += __shfl_xor(ls, off, 32);
      rs[r] = rs[r] * alpha + ls;
      #pragma unroll
      for (int nt = 0; nt < 4; ++nt) { aR[nt][r] *= alpha; aI[nt][r] *= alpha; }
    }

    // ---- P (C-layout) -> LDS -> A-fragments, then P @ V ----
    bf16* pw = &Ps[wid][0];
    #pragma unroll
    for (int kt = 0; kt < 4; ++kt)
      #pragma unroll
      for (int r = 0; r < 8; ++r)
        pw[(r + 8 * hf) * 64 + kt * 16 + l16] = (bf16)Sv[kt][r];

    AF pa[2];
    #pragma unroll
    for (int pf = 0; pf < 2; ++pf)
      #pragma unroll
      for (int v = 0; v < 8; ++v) {
        int kk = (v < 4) ? (hf * 8 + 2 * v) : (16 + hf * 8 + 2 * (v - 4));
        pa[pf].u[v] = *(const unsigned*)&pw[l16 * 64 + pf * 32 + kk];
      }

    #pragma unroll
    for (int nt = 0; nt < 4; ++nt)
      #pragma unroll
      for (int pf = 0; pf < 2; ++pf) {
        AF bvR, bvI;
        #pragma unroll
        for (int v = 0; v < 8; ++v) {
          int off = (nt * 16 + l16) * 64 + pf * 32 + ((v < 4) ? (hf * 8 + 2 * v) : (16 + hf * 8 + 2 * (v - 4)));
          bvR.u[v] = *(const unsigned*)&VsR[off];
          bvI.u[v] = *(const unsigned*)&VsI[off];
        }
        aR[nt] = wmma_bf16(pa[pf], bvR, aR[nt]);
        aI[nt] = wmma_bf16(pa[pf], bvI, aI[nt]);
      }
    __syncthreads();
  }

  // finalize: O / rowsum -> packed context (bf16)
  #pragma unroll
  for (int r = 0; r < 8; ++r) {
    float inv = 1.0f / rs[r];
    int tok = tb + q0 + wid * 16 + r + 8 * hf;
    bf16* crow = Cp + (size_t)tok * PK;
    #pragma unroll
    for (int nt = 0; nt < 4; ++nt) {
      crow[cr + nt * 16 + l16] = (bf16)(aR[nt][r] * inv);
      crow[ci + nt * 16 + l16] = (bf16)(aI[nt][r] * inv);
    }
  }
}

// ---------------------------------------------------------------------------
extern "C" void kernel_launch(void* const* d_in, const int* in_sizes, int n_in,
                              void* d_out, int out_size, void* d_ws, size_t ws_size,
                              hipStream_t stream) {
  (void)in_sizes; (void)n_in; (void)out_size; (void)ws_size;
  const float* x_r  = (const float*)d_in[0];
  const float* x_i  = (const float*)d_in[1];
  const float* mask = (const float*)d_in[2];
  const float *W_r[4], *W_i[4], *b_r[4], *b_i[4];
  for (int p = 0; p < 4; ++p) {            // q, k, v, o
    W_r[p] = (const float*)d_in[3 + 4 * p];
    W_i[p] = (const float*)d_in[4 + 4 * p];
    b_r[p] = (const float*)d_in[5 + 4 * p];
    b_i[p] = (const float*)d_in[6 + 4 * p];
  }

  char* ws = (char*)d_ws;
  const size_t SZ_X = (size_t)MTOK * PK * sizeof(bf16);  // 16 MiB
  const size_t SZ_W = (size_t)PK * PK * sizeof(bf16);    //  8 MiB
  bf16* Xp = (bf16*)ws;                                  // reused as context
  bf16* Wt[4];
  for (int p = 0; p < 4; ++p) Wt[p] = (bf16*)(ws + SZ_X + (size_t)p * SZ_W);
  bf16* Qp = (bf16*)(ws + SZ_X + 4 * SZ_W);
  bf16* Kp = Qp + (size_t)MTOK * PK;
  bf16* Vp = Kp + (size_t)MTOK * PK;
  bf16* Cp = Xp;                                         // alias (X dead after QKV)

  pack_x_kernel<<<4096, 256, 0, stream>>>(x_r, x_i, Xp);
  for (int p = 0; p < 4; ++p)
    pack_wT_kernel<<<2048, 256, 0, stream>>>(W_r[p], W_i[p], Wt[p]);

  dim3 gg(PK / 64, MTOK / 64);  // (32, 64)
  gemm64x64<0><<<gg, 128, 0, stream>>>(Xp, Wt[0], b_r[0], b_i[0], Qp);
  gemm64x64<0><<<gg, 128, 0, stream>>>(Xp, Wt[1], b_r[1], b_i[1], Kp);
  gemm64x64<0><<<gg, 128, 0, stream>>>(Xp, Wt[2], b_r[2], b_i[2], Vp);

  attn64<<<dim3(32, S_LEN / 64), 128, 0, stream>>>(Qp, Kp, Vp, mask, Cp);

  gemm64x64<1><<<gg, 128, 0, stream>>>(Cp, Wt[3], b_r[3], b_i[3], d_out);
}
